// LMheadWarpper_4303557231194
// MI455X (gfx1250) — matrix operand order repaired and checked by reading, not compiled, not run on hardware
//
#include <hip/hip_runtime.h>
#include <hip/hip_bf16.h>

// ---------------------------------------------------------------------------
// Fused LM-head + cross-entropy (sum) for MI455X / gfx1250, wave32.
//   logits = hidden @ weights^T   (N x V, K = D)  via v_wmma_f32_16x16x32_bf16
//   loss   = sum_rows ( logsumexp(logits[row]) - logits[row, label] )
// Weights pre-converted once to bf16 in d_ws (128 MB -> L2-resident; all
// blocks sweep V in the same order). Target logit computed in exact fp32.
// Per wave: 2 M-tiles x 2 N-tiles = 4 WMMAs per K-step, B double-buffered in
// registers so global-load latency overlaps the matrix ops.
// ---------------------------------------------------------------------------

typedef __attribute__((ext_vector_type(16))) __bf16        v16bf;
typedef __attribute__((ext_vector_type(8)))  float         v8f;
typedef __attribute__((ext_vector_type(4)))  float         v4f;
typedef __attribute__((ext_vector_type(2)))  unsigned int  v2u;
typedef __attribute__((ext_vector_type(4)))  unsigned int  v4u;
typedef __attribute__((ext_vector_type(8)))  unsigned int  v8u;

#define BM 32      // rows per block (2 WMMA M-tiles)
#define TN 2       // N-tiles per wave
#define BN 256     // cols per iteration (8 waves x 2 tiles x 16)
#define NTHREADS 256

// fp32 -> bf16 round-to-nearest-even (pure integer, always compiles)
__device__ __forceinline__ unsigned short f2bf(float x) {
  unsigned int u = __builtin_bit_cast(unsigned int, x);
  u += 0x7FFFu + ((u >> 16) & 1u);
  return (unsigned short)(u >> 16);
}

__device__ __forceinline__ v8u pack16(const float* q) {
  v4f f0 = *(const v4f*)q, f1 = *(const v4f*)(q + 4);
  v4f f2 = *(const v4f*)(q + 8), f3 = *(const v4f*)(q + 12);
  v8u bw;
  bw[0] = (unsigned)f2bf(f0[0]) | ((unsigned)f2bf(f0[1]) << 16);
  bw[1] = (unsigned)f2bf(f0[2]) | ((unsigned)f2bf(f0[3]) << 16);
  bw[2] = (unsigned)f2bf(f1[0]) | ((unsigned)f2bf(f1[1]) << 16);
  bw[3] = (unsigned)f2bf(f1[2]) | ((unsigned)f2bf(f1[3]) << 16);
  bw[4] = (unsigned)f2bf(f2[0]) | ((unsigned)f2bf(f2[1]) << 16);
  bw[5] = (unsigned)f2bf(f2[2]) | ((unsigned)f2bf(f2[3]) << 16);
  bw[6] = (unsigned)f2bf(f3[0]) | ((unsigned)f2bf(f3[1]) << 16);
  bw[7] = (unsigned)f2bf(f3[2]) | ((unsigned)f2bf(f3[3]) << 16);
  return bw;
}

// -------------------- weights fp32 -> bf16 pre-pass ------------------------
__global__ __launch_bounds__(NTHREADS)
void convert_w_bf16(const float* __restrict__ w, unsigned short* __restrict__ o,
                    size_t n8 /* n/8 */) {
  size_t i = (size_t)blockIdx.x * NTHREADS + threadIdx.x;
  if (i >= n8) return;
  size_t e = i * 8;
  v4f f0 = *(const v4f*)(w + e);
  v4f f1 = *(const v4f*)(w + e + 4);
  v4u pk;
  pk[0] = (unsigned)f2bf(f0[0]) | ((unsigned)f2bf(f0[1]) << 16);
  pk[1] = (unsigned)f2bf(f0[2]) | ((unsigned)f2bf(f0[3]) << 16);
  pk[2] = (unsigned)f2bf(f1[0]) | ((unsigned)f2bf(f1[1]) << 16);
  pk[3] = (unsigned)f2bf(f1[2]) | ((unsigned)f2bf(f1[3]) << 16);
  *(v4u*)(o + e) = pk;
}

// online logsumexp update, elementwise (no cross-lane work in hot path)
__device__ __forceinline__ void lse_upd(float& m, float& s, float x) {
  float mo = m, mn = fmaxf(mo, x);
  s = s * __expf(mo - mn) + __expf(x - mn);
  m = mn;
}

// ------------------------------ main kernel --------------------------------
template <bool PRECONV>
__global__ __launch_bounds__(NTHREADS)
void lmhead_ce_kernel(const float* __restrict__ hidden,
                      const long long* __restrict__ labels,
                      const float* __restrict__ weights,
                      const unsigned short* __restrict__ wbf,
                      float* __restrict__ out,
                      int N, int D, int V) {
  extern __shared__ unsigned char smem[];
  unsigned short* sA = (unsigned short*)smem;          // [BM][D+8] bf16
  const int tid     = threadIdx.x;
  const int row0    = blockIdx.x * BM;
  const int strideA = D + 8;                            // pad: kills bank conflicts
  const int Dq      = D >> 2;

  // ---- stage A tile (BM x D) into LDS as bf16, row-major ----
  for (int r = 0; r < BM; ++r) {
    for (int k4 = tid; k4 < Dq; k4 += NTHREADS) {
      v4f f = *(const v4f*)(hidden + (size_t)(row0 + r) * D + k4 * 4);
      v2u pk;
      pk[0] = (unsigned)f2bf(f[0]) | ((unsigned)f2bf(f[1]) << 16);
      pk[1] = (unsigned)f2bf(f[2]) | ((unsigned)f2bf(f[3]) << 16);
      *(v2u*)(sA + r * strideA + k4 * 4) = pk;
    }
  }
  __syncthreads();

  const int wave  = tid >> 5;
  const int lane  = tid & 31;
  const int mrow  = lane & 15;
  const int khalf = lane >> 4;
  const int KS    = D >> 5;

  // online-softmax state: per lane, 16 row-slots (2 tiles x 8 accum regs);
  // both of this lane's column classes fold into the same state.
  float m[16], s[16];
#pragma unroll
  for (int j = 0; j < 16; ++j) { m[j] = -__builtin_inff(); s[j] = 0.f; }

  const unsigned short* a0base = sA + mrow * strideA + khalf * 16;
  const unsigned short* a1base = a0base + 16 * strideA;

  const int nIter = V / BN;
  for (int it = 0; it < nIter; ++it) {
    // this lane's two B columns (B layout: lanes 0-15 hold K=0..15 contiguous)
    const int col0 = it * BN + wave * (16 * TN) + mrow;
    const size_t bOff0 = (size_t)col0 * D + khalf * 16;
    const size_t bOff1 = bOff0 + (size_t)16 * D;        // col0 + 16

    v8f acc00 = {}, acc01 = {}, acc10 = {}, acc11 = {};

    // prime B double-buffer with ks = 0
    v8u b0, b1;
    if (PRECONV) {
      b0 = *(const v8u*)(wbf + bOff0);
      b1 = *(const v8u*)(wbf + bOff1);
    } else {
      b0 = pack16(weights + bOff0);
      b1 = pack16(weights + bOff1);
    }

    for (int ks = 0; ks < KS; ++ks) {
      const int kb = ks * 32;
      const int kn = (ks + 1 < KS) ? kb + 32 : 0;       // harmless reload at end
      // prefetch next-step B into the shadow buffer (overlaps WMMAs below)
      v8u nb0, nb1;
      if (PRECONV) {
        nb0 = *(const v8u*)(wbf + bOff0 + kn);
        nb1 = *(const v8u*)(wbf + bOff1 + kn);
      } else {
        nb0 = pack16(weights + bOff0 + kn);
        nb1 = pack16(weights + bOff1 + kn);
      }
      // A tiles for this K-step (row mrow, K = kb+khalf*16 .. +15 contiguous)
      const unsigned short* a0p = a0base + kb;
      v4u a0l = *(const v4u*)a0p;
      v4u a0h = *(const v4u*)(a0p + 8);
      v16bf a0 = __builtin_bit_cast(v16bf,
          __builtin_shufflevector(a0l, a0h, 0, 1, 2, 3, 4, 5, 6, 7));
      const unsigned short* a1p = a1base + kb;
      v4u a1l = *(const v4u*)a1p;
      v4u a1h = *(const v4u*)(a1p + 8);
      v16bf a1 = __builtin_bit_cast(v16bf,
          __builtin_shufflevector(a1l, a1h, 0, 1, 2, 3, 4, 5, 6, 7));

      const v16bf bb0 = __builtin_bit_cast(v16bf, b0);
      const v16bf bb1 = __builtin_bit_cast(v16bf, b1);
      acc00 = __builtin_amdgcn_wmma_f32_16x16x32_bf16(
          false, a0, false, bb0, (short)0, acc00, false, false);
      acc01 = __builtin_amdgcn_wmma_f32_16x16x32_bf16(
          false, a1, false, bb0, (short)0, acc01, false, false);
      acc10 = __builtin_amdgcn_wmma_f32_16x16x32_bf16(
          false, a0, false, bb1, (short)0, acc10, false, false);
      acc11 = __builtin_amdgcn_wmma_f32_16x16x32_bf16(
          false, a1, false, bb1, (short)0, acc11, false, false);

      b0 = nb0;
      b1 = nb1;
    }

    // fold both column classes into the per-lane running logsumexp state
#pragma unroll
    for (int r = 0; r < 8; ++r) {
      lse_upd(m[r],     s[r],     acc00[r]);
      lse_upd(m[8 + r], s[8 + r], acc01[r]);
      lse_upd(m[r],     s[r],     acc10[r]);
      lse_upd(m[8 + r], s[8 + r], acc11[r]);
    }
  }

  // ---- merge per-lane partials across lanes & waves via LDS (alias sA) ----
  __syncthreads();
  float* pm = (float*)smem;            // [256][16]
  float* ps = pm + NTHREADS * 16;      // [256][16]
  float* tg = ps + NTHREADS * 16;      // [BM]
#pragma unroll
  for (int j = 0; j < 16; ++j) {
    pm[tid * 16 + j] = m[j];
    ps[tid * 16 + j] = s[j];
  }
  __syncthreads();

  // ---- exact fp32 target logit: 8 threads per row ----
  {
    const int trow = tid >> 3;
    const int kl   = tid & 7;
    long long lab  = labels[row0 + trow];
    float dot = 0.f;
    if (lab >= 0) {
      const int seg = D / 8;
      const float* hp = hidden  + (size_t)(row0 + trow) * D + kl * seg;
      const float* wp = weights + (size_t)lab * D + kl * seg;
      for (int k = 0; k < seg; k += 4) {
        v4f hv = *(const v4f*)(hp + k);
        v4f wv = *(const v4f*)(wp + k);
        dot = fmaf(hv[0], wv[0], dot);
        dot = fmaf(hv[1], wv[1], dot);
        dot = fmaf(hv[2], wv[2], dot);
        dot = fmaf(hv[3], wv[3], dot);
      }
    }
    dot += __shfl_xor(dot, 1, 32);
    dot += __shfl_xor(dot, 2, 32);
    dot += __shfl_xor(dot, 4, 32);
    if (kl == 0) tg[trow] = dot;
  }
  __syncthreads();

  // ---- final per-row logsumexp + nll, block sum, one atomic ----
  if (tid < BM) {
    const int R  = tid;
    const int t  = R >> 4;
    const int rr = R & 15;
    const int r  = rr & 7;
    const int hi = rr >> 3;
    const int j  = t * 8 + r;
    float mT = -__builtin_inff();
    for (int w = 0; w < 8; ++w)
      for (int c = 0; c < 16; ++c)
        mT = fmaxf(mT, pm[(w * 32 + hi * 16 + c) * 16 + j]);
    float sT = 0.f;
    for (int w = 0; w < 8; ++w)
      for (int c = 0; c < 16; ++c) {
        int th = (w * 32 + hi * 16 + c) * 16 + j;
        sT += ps[th] * __expf(pm[th] - mT);
      }
    long long lab = labels[row0 + R];
    float loss = (lab >= 0) ? (mT + __logf(sT) - tg[R]) : 0.f;
    loss += __shfl_xor(loss, 16, 32);
    loss += __shfl_xor(loss, 8, 32);
    loss += __shfl_xor(loss, 4, 32);
    loss += __shfl_xor(loss, 2, 32);
    loss += __shfl_xor(loss, 1, 32);
    if (tid == 0) atomicAdd(out, loss);
  }
}

// ------------------------------ launcher -----------------------------------
extern "C" void kernel_launch(void* const* d_in, const int* in_sizes, int n_in,
                              void* d_out, int out_size, void* d_ws, size_t ws_size,
                              hipStream_t stream) {
  const float*     hidden  = (const float*)d_in[0];
  const long long* labels  = (const long long*)d_in[1];
  const float*     weights = (const float*)d_in[2];
  float*           out     = (float*)d_out;

  const int N = in_sizes[1];
  const int D = in_sizes[0] / N;           // 2048
  const int V = in_sizes[2] / D;           // 32000

  hipMemsetAsync(out, 0, sizeof(float), stream);

  const size_t nW    = (size_t)V * (size_t)D;
  const size_t need  = nW * sizeof(unsigned short);    // 128 MB bf16 weights
  const size_t shmem = (size_t)BM * (size_t)(D + 8) * 2;
  const dim3 grid(N / BM), block(NTHREADS);

  if (ws_size >= need) {
    unsigned short* wbf = (unsigned short*)d_ws;
    const size_t n8 = nW / 8;
    const dim3 cgrid((unsigned)((n8 + NTHREADS - 1) / NTHREADS));
    convert_w_bf16<<<cgrid, block, 0, stream>>>(weights, wbf, n8);
    lmhead_ce_kernel<true><<<grid, block, shmem, stream>>>(
        hidden, labels, weights, wbf, out, N, D, V);
  } else {
    lmhead_ce_kernel<false><<<grid, block, shmem, stream>>>(
        hidden, labels, weights, nullptr, out, N, D, V);
  }
}
